// Attention_21320217657684
// MI455X (gfx1250) — matrix operand order repaired and checked by reading, not compiled, not run on hardware
//
#include <hip/hip_runtime.h>

// Problem constants (from reference): B=2, T=2048, D=4096, H=32, HKV=1, HD=128
#define Bc  2
#define Tc  2048
#define Dc  4096
#define Hc  32
#define HDc 128
#define Mc  (Bc * Tc)   // 4096 rows for all projection GEMMs

typedef __attribute__((ext_vector_type(16))) __bf16        v16bf;
typedef __attribute__((ext_vector_type(16))) unsigned short v16u;
typedef __attribute__((ext_vector_type(8)))  float          v8f;

union Frag16 { v16u u; v16bf bf; };

__device__ inline unsigned short f2bf(float f) {
  union { float f; unsigned u; } v; v.f = f;
  unsigned r = v.u + 0x7FFFu + ((v.u >> 16) & 1u);   // round-to-nearest-even
  return (unsigned short)(r >> 16);
}

__device__ inline v8f vzero8() {
  v8f v;
#pragma unroll
  for (int i = 0; i < 8; ++i) v[i] = 0.0f;
  return v;
}

__device__ inline v8f wmma_bf16(v16u a, v16u b, v8f c) {
  Frag16 fa, fb; fa.u = a; fb.u = b;
  // D = A(16x32 bf16) * B(32x16 bf16) + C(16x16 f32)
  return __builtin_amdgcn_wmma_f32_16x16x32_bf16(
      false, fa.bf, false, fb.bf, (short)0, c, false, false);
}

// ---- CDNA5 async global->LDS copy (ASYNCcnt path, CDNA5 ISA) ----
// 16 bytes per lane, memory -> LDS without VGPR round-trip.
// LDS operand = wave-relative LDS byte address = low 32 bits of generic shared ptr.
__device__ inline void async_copy_b128(const unsigned short* gsrc, unsigned short* ldst) {
  unsigned lds_off = (unsigned)(unsigned long long)ldst;
  asm volatile("global_load_async_to_lds_b128 %0, %1, off"
               :: "v"(lds_off), "v"(gsrc)
               : "memory");
}
__device__ inline void async_wait0() {
  asm volatile("s_wait_asynccnt 0x0" ::: "memory");
}

// A-fragment (16x32, MxK): lanes 0-15 hold K 0-15,
// lanes 16-31 hold K 16-31. src is row-major [rows][ld].
__device__ inline v16u frag_a(const unsigned short* src, int ld, int m0, int k0, int lane) {
  const unsigned short* p = src + (size_t)(m0 + (lane & 15)) * ld + k0 + ((lane >> 4) << 4);
  v16u a;
#pragma unroll
  for (int j = 0; j < 8; ++j) { a[j] = p[j]; a[8 + j] = p[8 + j]; }
  return a;
}

// B-fragment (32x16, KxN) from a [N][K] row-major tile (frag[k][n] = src[n0+n][k0+k]).
__device__ inline v16u frag_b(const unsigned short* src, int ld, int n0, int k0, int lane) {
  const unsigned short* p = src + (size_t)(n0 + (lane & 15)) * ld + k0 + ((lane >> 4) << 4);
  v16u b;
#pragma unroll
  for (int j = 0; j < 16; ++j) b[j] = p[j];
  return b;
}

// B-fragment (32x16, KxN) from a [K][N] row-major tile (frag[k][n] = src[k0+k][n0+n]).
__device__ inline v16u frag_b_km(const unsigned short* src, int ld, int k0, int n0, int lane) {
  const int n  = n0 + (lane & 15);
  const int kb = k0 + ((lane >> 4) << 4);
  v16u b;
#pragma unroll
  for (int j = 0; j < 16; ++j) b[j] = src[(size_t)(kb + j) * ld + n];
  return b;
}

// ---------------------------------------------------------------- fp32 -> bf16
__global__ void cast_f32_bf16_kernel(const float* __restrict__ src,
                                     unsigned short* __restrict__ dst, int n) {
  int i = blockIdx.x * blockDim.x + threadIdx.x;
  const int stride = gridDim.x * blockDim.x;
  for (; i < n; i += stride) dst[i] = f2bf(src[i]);
}

// ---------------------------------------------------------------- GEMM
// C[M,N] = alpha * (A[M,K] @ W[N,K]^T + bias[N]) ; A,W bf16 ; C fp32 or bf16.
// Tile: 128x128x32, 256 threads (8 waves), each wave -> 32x64 (2x4 WMMA tiles).
// Double-buffered LDS with async global->LDS staging.
#define GBM 128
#define GBN 128
#define GBK 32
#define GLD 40  // BK + 8 pad (keeps rows 16B-aligned, eases bank conflicts)

__device__ inline void gemm_stage(const unsigned short* __restrict__ A,
                                  const unsigned short* __restrict__ W,
                                  unsigned short* AsBuf, unsigned short* BsBuf,
                                  int m0, int n0, int K, int k0, int tid) {
#pragma unroll
  for (int i = 0; i < 2; ++i) {
    const int idx = tid + (i << 8);          // 0..511
    const int row = idx >> 2;                // 0..127
    const int c8  = (idx & 3) << 3;          // 0,8,16,24
    async_copy_b128(&A[(size_t)(m0 + row) * K + k0 + c8], &AsBuf[row * GLD + c8]);
    async_copy_b128(&W[(size_t)(n0 + row) * K + k0 + c8], &BsBuf[row * GLD + c8]);
  }
}

__global__ __launch_bounds__(256)
void gemm_bias_kernel(const unsigned short* __restrict__ A,
                      const unsigned short* __restrict__ W,
                      const float* __restrict__ bias,
                      float* __restrict__ Cf, unsigned short* __restrict__ Cb,
                      int M, int N, int K, float alpha) {
  __shared__ unsigned short As[2][GBM * GLD];
  __shared__ unsigned short Bs[2][GBN * GLD];

  const int tid  = threadIdx.x;
  const int lane = tid & 31;
  const int wid  = tid >> 5;
  const int wm   = (wid & 3) * 32;   // 4 waves tile M
  const int wn   = (wid >> 2) * 64;  // 2 waves tile N
  const int m0   = blockIdx.y * GBM;
  const int n0   = blockIdx.x * GBN;

  v8f acc[2][4];
#pragma unroll
  for (int i = 0; i < 2; ++i)
#pragma unroll
    for (int j = 0; j < 4; ++j) acc[i][j] = vzero8();

  const int nk = K / GBK;
  gemm_stage(A, W, As[0], Bs[0], m0, n0, K, 0, tid);   // prologue: tile 0 in flight

  for (int kt = 0; kt < nk; ++kt) {
    const int buf = kt & 1;
    async_wait0();        // my async copies for tile kt have landed
    __syncthreads();      // everyone's copies landed; prev tile fully consumed

    if (kt + 1 < nk)      // stage tile kt+1 into the other buffer
      gemm_stage(A, W, As[buf ^ 1], Bs[buf ^ 1], m0, n0, K, (kt + 1) * GBK, tid);

    const v16u af0 = frag_a(As[buf], GLD, wm,      0, lane);
    const v16u af1 = frag_a(As[buf], GLD, wm + 16, 0, lane);
#pragma unroll
    for (int j = 0; j < 4; ++j) {
      const v16u bfj = frag_b(Bs[buf], GLD, wn + j * 16, 0, lane);
      acc[0][j] = wmma_bf16(af0, bfj, acc[0][j]);
      acc[1][j] = wmma_bf16(af1, bfj, acc[1][j]);
    }
    __syncthreads();      // done reading buf before it is overwritten two tiles later
  }

  const int col = lane & 15;
  const int rb  = (lane >> 4) * 8;   // C layout: upper lane-half holds rows 8..15
#pragma unroll
  for (int i = 0; i < 2; ++i)
#pragma unroll
    for (int j = 0; j < 4; ++j) {
      const int n  = n0 + wn + j * 16 + col;
      const float bv = bias[n];
#pragma unroll
      for (int r = 0; r < 8; ++r) {
        const int m = m0 + wm + i * 16 + rb + r;
        const float v = alpha * (acc[i][j][r] + bv);
        if (Cf) Cf[(size_t)m * N + n] = v;
        else    Cb[(size_t)m * N + n] = f2bf(v);
      }
    }
}

// ---------------------------------------------------------------- Flash attention
// grid (T/64, H, B), 128 threads = 4 waves; each wave owns 16 query rows.
// K/V (HKV=1) streamed through LDS (async copies) in 64-key tiles; online softmax; causal.
#define ABM 64
#define ABN 64
#define KLD 136   // HD + 8 pad
#define PLD 72    // ABN + 8 pad

__global__ __launch_bounds__(128)
void flash_attn_kernel(const unsigned short* __restrict__ Q,
                       const unsigned short* __restrict__ Kmat,
                       const unsigned short* __restrict__ Vmat,
                       unsigned short* __restrict__ Ob) {
  __shared__ unsigned short Ks[ABN * KLD];
  __shared__ unsigned short Vs[ABN * KLD];
  __shared__ unsigned short Ps[4 * 16 * PLD];  // per-wave P staging

  const int tid   = threadIdx.x;
  const int lane  = tid & 31;
  const int wid   = tid >> 5;
  const int qTile = blockIdx.x * ABM;
  const int h     = blockIdx.y;
  const int b     = blockIdx.z;
  const int q0w   = qTile + wid * 16;
  const int col   = lane & 15;
  const int rb    = (lane >> 4) * 8;

  // Q fragments for this wave's 16 rows: 4 chunks of K=32 over HD=128 (pre-scaled by 1/sqrt(HD))
  v16u qf[4];
  {
    const unsigned short* qp =
        Q + (size_t)(b * Tc + q0w + (lane & 15)) * Dc + h * HDc;
    const int kh = (lane >> 4) << 4;
#pragma unroll
    for (int c = 0; c < 4; ++c) {
      const unsigned short* p = qp + c * 32 + kh;
#pragma unroll
      for (int j = 0; j < 8; ++j) { qf[c][j] = p[j]; qf[c][8 + j] = p[8 + j]; }
    }
  }

  v8f o[8];
#pragma unroll
  for (int i = 0; i < 8; ++i) o[i] = vzero8();
  float rmax[8], rsum[8];
#pragma unroll
  for (int r = 0; r < 8; ++r) { rmax[r] = -1e30f; rsum[r] = 0.0f; }

  const int kend = (qTile + ABM < Tc) ? (qTile + ABM) : Tc;
  for (int kb0 = 0; kb0 < kend; kb0 += ABN) {
    // stage K/V tiles (64 x 128) into LDS with async copies
#pragma unroll
    for (int i = 0; i < 8; ++i) {
      const int idx = tid + i * 128;       // 0..1023
      const int row = idx >> 4;            // 0..63
      const int c8  = (idx & 15) << 3;     // 0..120
      async_copy_b128(&Kmat[(size_t)(b * Tc + kb0 + row) * HDc + c8], &Ks[row * KLD + c8]);
      async_copy_b128(&Vmat[(size_t)(b * Tc + kb0 + row) * HDc + c8], &Vs[row * KLD + c8]);
    }
    async_wait0();
    __syncthreads();

    // S = Q @ K^T  (16 x 64), 16 WMMAs
    v8f s[4];
#pragma unroll
    for (int nt = 0; nt < 4; ++nt) s[nt] = vzero8();
#pragma unroll
    for (int nt = 0; nt < 4; ++nt)
#pragma unroll
      for (int c = 0; c < 4; ++c) {
        const v16u kf = frag_b(Ks, KLD, nt * 16, c * 32, lane);
        s[nt] = wmma_bf16(qf[c], kf, s[nt]);
      }

    // causal mask + per-row block max (rows live in VGPR index, keys across lanes)
    float bmax[8];
#pragma unroll
    for (int r = 0; r < 8; ++r) bmax[r] = -1e30f;
#pragma unroll
    for (int nt = 0; nt < 4; ++nt) {
      const int kg = kb0 + nt * 16 + col;
#pragma unroll
      for (int r = 0; r < 8; ++r) {
        const int qg = q0w + rb + r;
        float v = s[nt][r];
        if (kg > qg) v = -1e30f;
        s[nt][r] = v;
        bmax[r] = fmaxf(bmax[r], v);
      }
    }
#pragma unroll
    for (int msk = 1; msk < 16; msk <<= 1)
#pragma unroll
      for (int r = 0; r < 8; ++r)
        bmax[r] = fmaxf(bmax[r], __shfl_xor(bmax[r], msk, 32));

    // online softmax update
    unsigned short* myP = &Ps[wid * 16 * PLD];
#pragma unroll
    for (int r = 0; r < 8; ++r) {
      const float nm   = fmaxf(rmax[r], bmax[r]);
      const float corr = __expf(rmax[r] - nm);
      rmax[r] = nm;
      float ps = 0.0f;
#pragma unroll
      for (int nt = 0; nt < 4; ++nt) {
        const float p = __expf(s[nt][r] - nm);
        s[nt][r] = p;
        ps += p;
      }
#pragma unroll
      for (int msk = 1; msk < 16; msk <<= 1) ps += __shfl_xor(ps, msk, 32);
      rsum[r] = rsum[r] * corr + ps;
#pragma unroll
      for (int dt = 0; dt < 8; ++dt) o[dt][r] *= corr;
    }

    // re-fragment P: C-layout -> LDS -> A-layout (same-wave, compiler waits DScnt)
#pragma unroll
    for (int nt = 0; nt < 4; ++nt)
#pragma unroll
      for (int r = 0; r < 8; ++r)
        myP[(size_t)(rb + r) * PLD + nt * 16 + col] = f2bf(s[nt][r]);

    // O += P @ V  (16 x 128), 16 WMMAs
#pragma unroll
    for (int kc = 0; kc < 2; ++kc) {
      const v16u pf = frag_a(myP, PLD, 0, kc * 32, lane);
#pragma unroll
      for (int dt = 0; dt < 8; ++dt) {
        const v16u vf = frag_b_km(Vs, KLD, kc * 32, dt * 16, lane);
        o[dt] = wmma_bf16(pf, vf, o[dt]);
      }
    }
    __syncthreads();   // all waves done with Ks/Vs before next stage is issued
  }

  // normalize and store bf16 attention output (B,T,H*HD)
#pragma unroll
  for (int dt = 0; dt < 8; ++dt)
#pragma unroll
    for (int r = 0; r < 8; ++r) {
      const int qg = q0w + rb + r;
      const float v = o[dt][r] / rsum[r];
      Ob[(size_t)(b * Tc + qg) * Dc + h * HDc + dt * 16 + col] = f2bf(v);
    }
}

// ---------------------------------------------------------------- host
extern "C" void kernel_launch(void* const* d_in, const int* in_sizes, int n_in,
                              void* d_out, int out_size, void* d_ws, size_t ws_size,
                              hipStream_t stream) {
  (void)in_sizes; (void)n_in; (void)out_size; (void)ws_size;

  const float* x    = (const float*)d_in[0];
  const float* wq_w = (const float*)d_in[1];
  const float* wq_b = (const float*)d_in[2];
  const float* wk_w = (const float*)d_in[3];
  const float* wk_b = (const float*)d_in[4];
  const float* wv_w = (const float*)d_in[5];
  const float* wv_b = (const float*)d_in[6];
  const float* wo_w = (const float*)d_in[7];
  const float* wo_b = (const float*)d_in[8];
  float* out = (float*)d_out;

  // workspace carve (bf16 buffers), ~164 MB total
  unsigned char* ws = (unsigned char*)d_ws;
  size_t off = 0;
  auto carve = [&](size_t elems) {
    unsigned short* p = (unsigned short*)(ws + off);
    off += elems * sizeof(unsigned short);
    return p;
  };
  unsigned short* xb  = carve((size_t)Mc * Dc);    // x bf16
  unsigned short* wqb = carve((size_t)Dc * Dc);    // wq bf16
  unsigned short* wkb = carve((size_t)HDc * Dc);   // wk bf16
  unsigned short* wvb = carve((size_t)HDc * Dc);   // wv bf16
  unsigned short* wob = carve((size_t)Dc * Dc);    // wo bf16
  unsigned short* qb  = carve((size_t)Mc * Dc);    // q (pre-scaled) bf16
  unsigned short* kb  = carve((size_t)Mc * HDc);   // k bf16
  unsigned short* vb  = carve((size_t)Mc * HDc);   // v bf16
  unsigned short* ab  = carve((size_t)Mc * Dc);    // attn out bf16

  cast_f32_bf16_kernel<<<2048, 256, 0, stream>>>(x,    xb,  Mc * Dc);
  cast_f32_bf16_kernel<<<2048, 256, 0, stream>>>(wq_w, wqb, Dc * Dc);
  cast_f32_bf16_kernel<<<64,   256, 0, stream>>>(wk_w, wkb, HDc * Dc);
  cast_f32_bf16_kernel<<<64,   256, 0, stream>>>(wv_w, wvb, HDc * Dc);
  cast_f32_bf16_kernel<<<2048, 256, 0, stream>>>(wo_w, wob, Dc * Dc);

  const float qscale = 0.08838834764831845f;  // 1/sqrt(HD)

  gemm_bias_kernel<<<dim3(Dc / GBN, Mc / GBM), 256, 0, stream>>>(
      xb, wqb, wq_b, nullptr, qb, Mc, Dc, Dc, qscale);
  gemm_bias_kernel<<<dim3(1, Mc / GBM), 256, 0, stream>>>(
      xb, wkb, wk_b, nullptr, kb, Mc, HDc, Dc, 1.0f);
  gemm_bias_kernel<<<dim3(1, Mc / GBM), 256, 0, stream>>>(
      xb, wvb, wv_b, nullptr, vb, Mc, HDc, Dc, 1.0f);

  flash_attn_kernel<<<dim3(Tc / ABM, Hc, Bc), 128, 0, stream>>>(qb, kb, vb, ab);

  gemm_bias_kernel<<<dim3(Dc / GBN, Mc / GBM), 256, 0, stream>>>(
      ab, wob, wo_b, out, nullptr, Mc, Dc, Dc, 1.0f);
}
